// TypeAwareDeconfoundedGAT_18202071400570
// MI455X (gfx1250) — matrix operand order repaired and checked by reading, not compiled, not run on hardware
//
#include <hip/hip_runtime.h>

#define NN 50000      // nodes
#define NPAD 50048    // nodes padded to 391*128 rows for tail-safe loads
#define NE 800000     // edges
#define CIN 128       // in channels
#define DOUT 128      // H*C
#define HEADS 4
#define CH 32         // per-head channels
#define NT 4          // edge types
#define NK 5          // confounders

typedef __attribute__((ext_vector_type(16))) __bf16 v16bf;
typedef __attribute__((ext_vector_type(8)))  __bf16 v8bf;
typedef __attribute__((ext_vector_type(8)))  float  v8f;

__device__ __forceinline__ __bf16 to_bf16(float f) {
  unsigned u = __float_as_uint(f);
  unsigned r = (u + 0x7FFFu + ((u >> 16) & 1u)) >> 16;   // round-to-nearest-even
  unsigned short s = (unsigned short)r;
  __bf16 h;
  __builtin_memcpy(&h, &s, sizeof(h));
  return h;
}

// order-preserving float <-> uint mapping for atomicMax on floats
__device__ __forceinline__ unsigned f2ord(float f) {
  unsigned u = __float_as_uint(f);
  return (u & 0x80000000u) ? ~u : (u | 0x80000000u);
}
__device__ __forceinline__ float ord2f(unsigned u) {
  return (u & 0x80000000u) ? __uint_as_float(u & 0x7fffffffu) : __uint_as_float(~u);
}

// ---------------------------------------------------------------------------
// Pre-pass A: x (f32) -> x_bf (bf16), zero-padded to NPAD rows
// ---------------------------------------------------------------------------
__global__ void conv_x_kernel(const float* __restrict__ x, __bf16* __restrict__ xb)
{
  int i = blockIdx.x * blockDim.x + threadIdx.x;
  if (i >= NPAD * CIN) return;
  float v = (i < NN * CIN) ? x[i] : 0.0f;
  xb[i] = to_bf16(v);
}

// ---------------------------------------------------------------------------
// Pre-pass B: W (f32) -> fragment-shaped bf16 pack:
//   wp[((t*4+ks)*8+nt)*32*16 + lane*16 + e] = W[t][ks*32 + (lane>>4)*8 + (e&7) + (e>>3)*16][nt*16 + (lane&15)]
// so each lane's 16-element B fragment is one contiguous 32B run.
// ---------------------------------------------------------------------------
__global__ void pack_w_kernel(const float* __restrict__ W, __bf16* __restrict__ wp)
{
  int i = blockIdx.x * blockDim.x + threadIdx.x;   // 65536 total
  const int e  = i & 15;
  const int l  = (i >> 4) & 31;
  const int nt = (i >> 9) & 7;
  const int ks = (i >> 12) & 3;
  const int t  = i >> 14;
  const int k = ks * 32 + (l >> 4) * 8 + (e & 7) + (e >> 3) * 16;
  const int n = nt * 16 + (l & 15);
  wp[i] = to_bf16(W[(size_t)t * CIN * DOUT + (size_t)k * DOUT + n]);
}

// ---------------------------------------------------------------------------
// Pass 1: xt[t] = x @ W[t] via bf16 WMMA, fp32 accumulate.
// grid = (391, 4), block = 256 (8 waves). Each wave: one 16-row M-tile,
// all 8 N-tiles (8 accumulators), 32 v_wmma per wave, A fragment reused 8x.
// ---------------------------------------------------------------------------
__global__ void __launch_bounds__(256)
gemm_xt_kernel(const __bf16* __restrict__ xb, const __bf16* __restrict__ wp,
               float* __restrict__ xt)
{
  const int t    = blockIdx.y;
  const int wave = threadIdx.x >> 5;
  const int lane = threadIdx.x & 31;
  const int half = lane >> 4;
  const int lr   = lane & 15;
  const int mt   = blockIdx.x * 8 + wave;          // M tile (16 rows); may be tail

  const __bf16* __restrict__ xrow = xb + (size_t)(mt * 16 + lr) * CIN;
  const __bf16* __restrict__ wt   = wp + (size_t)t * (4 * 8 * 32 * 16) + lane * 16;

  v8f acc[8] = {};
#pragma unroll
  for (int ks = 0; ks < 4; ++ks) {
    // A fragment: two contiguous 8-bf16 (16B) runs per lane
    const int ka = ks * 32 + half * 8;
    const v8bf a0 = *(const v8bf*)(xrow + ka);
    const v8bf a1 = *(const v8bf*)(xrow + ka + 16);
    const v16bf a = __builtin_shufflevector(a0, a1,
        0, 1, 2, 3, 4, 5, 6, 7, 8, 9, 10, 11, 12, 13, 14, 15);
#pragma unroll
    for (int nt = 0; nt < 8; ++nt) {
      // B fragment: one contiguous 32B run per lane (L0-resident, shared by waves)
      const v16bf b = *(const v16bf*)(wt + ((size_t)ks * 8 + nt) * (32 * 16));
      acc[nt] = __builtin_amdgcn_wmma_f32_16x16x32_bf16(
          false, a, false, b, (short)0, acc[nt], false, false);
    }
  }

  if (mt >= NN / 16) return;                        // tail guard (stores only)
  float* __restrict__ base =
      xt + ((size_t)t * NN + (size_t)mt * 16) * DOUT + lr;
#pragma unroll
  for (int nt = 0; nt < 8; ++nt)
#pragma unroll
    for (int g = 0; g < 8; ++g)
      base[(size_t)(half * 8 + g) * DOUT + nt * 16] = acc[nt][g];
}

// ---------------------------------------------------------------------------
// Init kernels
// ---------------------------------------------------------------------------
__global__ void init_seg_kernel(unsigned* __restrict__ mbits, float* __restrict__ denom)
{
  int i = blockIdx.x * blockDim.x + threadIdx.x;
  if (i < NT * NN * HEADS) {
    mbits[i] = f2ord(-3.0e38f);
    denom[i] = 0.0f;
  }
}

__global__ void init_out_kernel(const float* __restrict__ xt0,
                                const float* __restrict__ bias,
                                float* __restrict__ out)
{
  int i = blockIdx.x * blockDim.x + threadIdx.x;
  if (i < NN * DOUT) out[i] = xt0[i] + bias[i & (DOUT - 1)];
}

// ---------------------------------------------------------------------------
// Pass 2: per-edge attention logit alpha[e,h] + segment max. one wave per edge.
// ---------------------------------------------------------------------------
__global__ void __launch_bounds__(256)
edge_alpha_kernel(const float* __restrict__ xt, const int* __restrict__ ei,
                  const int* __restrict__ et, const float* __restrict__ att,
                  const float* __restrict__ cprob,
                  float* __restrict__ alpha, unsigned* __restrict__ mbits)
{
  const int e = (blockIdx.x * blockDim.x + threadIdx.x) >> 5;
  const int lane = threadIdx.x & 31;
  if (e >= NE) return;

  const int t = et[e];
  const int s = ei[e];        // src (message j)
  const int d = ei[NE + e];   // dst (message i)

  const float4* __restrict__ xi = (const float4*)(xt + ((size_t)t * NN + d) * DOUT);
  const float4* __restrict__ xj = (const float4*)(xt + ((size_t)t * NN + s) * DOUT);
  const float4 vi = xi[lane];
  const float4 vj = xj[lane];

  const int h  = lane >> 3;          // 8 lanes per head
  const int cc = (lane & 7) * 4;     // channel within head (0..28)

  // softmax over confounder priors (5 values, recomputed locally)
  float p[NK], mx = -3.0e38f, ssum = 0.0f;
#pragma unroll
  for (int k = 0; k < NK; ++k) { p[k] = cprob[k]; mx = fmaxf(mx, p[k]); }
#pragma unroll
  for (int k = 0; k < NK; ++k) { p[k] = __expf(p[k] - mx); ssum += p[k]; }

  float alpha_h = 0.0f;
#pragma unroll
  for (int k = 0; k < NK; ++k) {
    const float* __restrict__ av = att + (size_t)(k * HEADS + h) * (2 * CH);
    float part = vi.x * av[cc]          + vi.y * av[cc + 1]
               + vi.z * av[cc + 2]      + vi.w * av[cc + 3]
               + vj.x * av[CH + cc]     + vj.y * av[CH + cc + 1]
               + vj.z * av[CH + cc + 2] + vj.w * av[CH + cc + 3];
    part += __shfl_xor(part, 1, 32);
    part += __shfl_xor(part, 2, 32);
    part += __shfl_xor(part, 4, 32);           // full per-(h,k) score in 8-lane group
    float sc = (part > 0.0f) ? part : 0.2f * part;   // leaky relu
    alpha_h += (p[k] / ssum) * sc;
  }

  if ((lane & 7) == 0) {
    alpha[(size_t)e * HEADS + h] = alpha_h;
    const size_t seg = (size_t)t * NN + d;
    atomicMax(&mbits[seg * HEADS + h], f2ord(alpha_h));
  }
}

// ---------------------------------------------------------------------------
// Pass 3: a = exp(alpha - m); denom[seg,h] += a. one thread per (edge, head).
// ---------------------------------------------------------------------------
__global__ void __launch_bounds__(256)
edge_denom_kernel(float* __restrict__ alpha, const unsigned* __restrict__ mbits,
                  const int* __restrict__ ei, const int* __restrict__ et,
                  float* __restrict__ denom)
{
  int i = blockIdx.x * blockDim.x + threadIdx.x;
  if (i >= NE * HEADS) return;
  const int e = i >> 2;
  const int h = i & 3;
  const size_t seg = (size_t)et[e] * NN + ei[NE + e];
  const float a = __expf(alpha[i] - ord2f(mbits[seg * HEADS + h]));
  alpha[i] = a;                        // overwrite logit with numerator
  atomicAdd(&denom[seg * HEADS + h], a);
}

// ---------------------------------------------------------------------------
// Pass 4: scatter messages: out[dst] += xj * (a/denom) * importance[t].
// one wave per edge; lane owns 4 contiguous channels (one head per 8 lanes).
// ---------------------------------------------------------------------------
__global__ void __launch_bounds__(256)
edge_scatter_kernel(const float* __restrict__ xt, const float* __restrict__ expa,
                    const float* __restrict__ denom, const int* __restrict__ ei,
                    const int* __restrict__ et, const float* __restrict__ imp,
                    float* __restrict__ out)
{
  const int e = (blockIdx.x * blockDim.x + threadIdx.x) >> 5;
  const int lane = threadIdx.x & 31;
  if (e >= NE) return;

  const int t = et[e];
  const int s = ei[e];
  const int d = ei[NE + e];
  const int h = lane >> 3;

  const size_t seg = (size_t)t * NN + d;
  const float w = expa[(size_t)e * HEADS + h] / denom[seg * HEADS + h] * imp[t];

  const float4* __restrict__ xj = (const float4*)(xt + ((size_t)t * NN + s) * DOUT);
  const float4 v = xj[lane];
  float* __restrict__ o = out + (size_t)d * DOUT + lane * 4;
  atomicAdd(o + 0, v.x * w);
  atomicAdd(o + 1, v.y * w);
  atomicAdd(o + 2, v.z * w);
  atomicAdd(o + 3, v.w * w);
}

// ---------------------------------------------------------------------------
// Host launcher
// ---------------------------------------------------------------------------
extern "C" void kernel_launch(void* const* d_in, const int* in_sizes, int n_in,
                              void* d_out, int out_size, void* d_ws, size_t ws_size,
                              hipStream_t stream) {
  const float* x     = (const float*)d_in[0];
  const int*   ei    = (const int*)  d_in[1];   // [2, E]: src then dst
  const int*   et    = (const int*)  d_in[2];
  const float* W     = (const float*)d_in[3];   // [4,128,128]
  const float* att   = (const float*)d_in[4];   // [5,4,64]
  const float* cpr   = (const float*)d_in[5];   // [5]
  const float* imp   = (const float*)d_in[6];   // [4]
  const float* bias  = (const float*)d_in[7];   // [128]
  float* out = (float*)d_out;

  // workspace layout (bytes, all 256B-aligned section starts)
  char* ws = (char*)d_ws;
  float*    xt    = (float*)(ws);                                       // 102.4 MB
  float*    alpha = (float*)(ws + (size_t)NT * NN * DOUT * 4);          // 12.8 MB
  unsigned* mbits = (unsigned*)((char*)alpha + (size_t)NE * HEADS * 4); // 3.2 MB
  float*    denom = (float*)((char*)mbits + (size_t)NT * NN * HEADS * 4); // 3.2 MB
  __bf16*   xbf   = (__bf16*)((char*)denom + (size_t)NT * NN * HEADS * 4); // 12.8 MB
  __bf16*   wpk   = (__bf16*)((char*)xbf + (size_t)NPAD * CIN * 2);       // 128 KB

  // 0) one-time-per-call conversions / packing
  conv_x_kernel<<<(NPAD * CIN) / 256, 256, 0, stream>>>(x, xbf);
  pack_w_kernel<<<(NT * CIN * DOUT) / 256, 256, 0, stream>>>(W, wpk);

  // 1) segment-state init (max -> -inf ordinal, denom -> 0)
  init_seg_kernel<<<(NT * NN * HEADS) / 256, 256, 0, stream>>>(mbits, denom);

  // 2) typed transform xt[t] = x @ W[t] via bf16 WMMA
  gemm_xt_kernel<<<dim3(NPAD / 128, NT), 256, 0, stream>>>(xbf, wpk, xt);

  // 3) out = xt[0] + bias  (self-loop term, reuses GEMM result)
  init_out_kernel<<<(NN * DOUT) / 256, 256, 0, stream>>>(xt, bias, out);

  // 4) attention logits + segment max
  edge_alpha_kernel<<<NE / 8, 256, 0, stream>>>(xt, ei, et, att, cpr, alpha, mbits);

  // 5) exp + segment sum
  edge_denom_kernel<<<(NE * HEADS) / 256, 256, 0, stream>>>(alpha, mbits, ei, et, denom);

  // 6) normalized weighted scatter
  edge_scatter_kernel<<<NE / 8, 256, 0, stream>>>(xt, alpha, denom, ei, et, imp, out);
}
